// ContrastiveLoss_6528350290538
// MI455X (gfx1250) — matrix operand order repaired and checked by reading, not compiled, not run on hardware
//
#include <hip/hip_runtime.h>

// Problem dimensions (from reference setup_inputs)
#define N_  512
#define D_  256
#define T_  128
#define K_  (D_ * T_)   // 32768 contraction length

#define LDSROW 72       // 64 halves + 8 pad -> 144B row stride, de-conflicts LDS banks

// Fast path tiling: 64x64 tiles, split-K for occupancy (8x8x8 = 512 blocks)
#define SPLITK 8
#define KSLICE (K_ / SPLITK)  // 4096
#define FBM 64
#define FBN 64
#define FBK 64

// Fallback tiling (round-1 kernel, ~1.3MB workspace)
#define BM 128
#define BN 128
#define BK 64

typedef __attribute__((ext_vector_type(4)))  _Float16 v4h;
typedef __attribute__((ext_vector_type(8)))  _Float16 v8h;
typedef __attribute__((ext_vector_type(16))) _Float16 v16h;
typedef __attribute__((ext_vector_type(4)))  float    v4f;
typedef __attribute__((ext_vector_type(8)))  float    v8f;
typedef __attribute__((ext_vector_type(4)))  int      v4i;

// ---------------------------------------------------------------------------
// gfx1250 async global->LDS path (ASYNCcnt), with safe fallback.
// Builtin signature (from compiler diagnostic): (int4 AS1*, int4 AS3*, imm, imm)
// ---------------------------------------------------------------------------
#if defined(__AMDGCN__) && __has_builtin(__builtin_amdgcn_global_load_async_to_lds_b128)
#define HAVE_ASYNC_LDS 1
#else
#define HAVE_ASYNC_LDS 0
#endif

__device__ __forceinline__ void copy16_to_lds(const _Float16* g, _Float16* l) {
#if HAVE_ASYNC_LDS
    __builtin_amdgcn_global_load_async_to_lds_b128(
        (__attribute__((address_space(1))) v4i*)g,
        (__attribute__((address_space(3))) v4i*)l, 0, 0);
#else
    *(v8h*)l = *(const v8h*)g;
#endif
}

__device__ __forceinline__ void async_join() {
#if HAVE_ASYNC_LDS
#if __has_builtin(__builtin_amdgcn_s_wait_asynccnt)
    __builtin_amdgcn_s_wait_asynccnt(0);
#else
    asm volatile("s_wait_asynccnt 0x0" ::: "memory");
#endif
#endif
}

// ---------------------------------------------------------------------------
// FAST PATH kernel 1: fused norm + fp32->f16 normalized copy.
// One block per n; threads split D in half to compute column norms, then the
// whole block converts the 32768-element slice. fnh is 32MB -> L2 resident.
// ---------------------------------------------------------------------------
__global__ __launch_bounds__(256)
void fnh_kernel(const float* __restrict__ feats, _Float16* __restrict__ fnh) {
    __shared__ float part[256];
    __shared__ float invn[T_];
    const int n   = blockIdx.x;
    const int tid = threadIdx.x;
    const int t   = tid & (T_ - 1);
    const int dh  = tid >> 7;                 // 0/1: even/odd d
    const float* base = feats + (size_t)n * K_;

    float acc = 0.0f;
#pragma unroll 4
    for (int j = 0; j < D_ / 2; ++j) {
        float v = base[(size_t)(2 * j + dh) * T_ + t];
        acc += v * v;
    }
    part[tid] = acc;
    __syncthreads();
    if (tid < T_) {
        float s = part[tid] + part[tid + T_];
        invn[tid] = 1.0f / fmaxf(sqrtf(s), 1e-8f);
    }
    __syncthreads();

#pragma unroll
    for (int i = 0; i < 32; ++i) {
        const int k = (tid + i * 256) * 4;    // 0..32764, step 4
        v4f f = *(const v4f*)(base + k);
        const int tk = k & (T_ - 1);          // t index of first lane-element
        v4h h;
#pragma unroll
        for (int c = 0; c < 4; ++c) h[c] = (_Float16)(f[c] * invn[tk + c]);
        *(v4h*)(fnh + (size_t)n * K_ + k) = h;
    }
}

// ---------------------------------------------------------------------------
// FAST PATH kernel 2: split-K Gram partials. Grid (8,8,SPLITK) = 512 blocks,
// 128 threads (4 waves), 64x64 tile, double-buffered LDS staged with async
// b128 copies; each wave computes a 32x32 tile = 2x2 WMMA accumulators.
// ---------------------------------------------------------------------------
__global__ __launch_bounds__(128)
void gram_f16_kernel(const _Float16* __restrict__ fnh, float* __restrict__ partial) {
    __shared__ __align__(16) _Float16 As[2][FBM * LDSROW];
    __shared__ __align__(16) _Float16 Bs[2][FBN * LDSROW];

    const int tid  = threadIdx.x;
    const int lane = tid & 31;
    const int wave = tid >> 5;          // 0..3
    const int wrow = wave >> 1;         // 0..1 : 32-row strip
    const int wcol = wave & 1;          // 0..1 : 32-col strip
    const int lrow = lane & 15;
    const int hsel = (lane < 16) ? 0 : 1;
    const int bm = blockIdx.y * FBM;
    const int bn = blockIdx.x * FBN;
    const int k0 = blockIdx.z * KSLICE;

    v8f acc[2][2];
#pragma unroll
    for (int mi = 0; mi < 2; ++mi)
#pragma unroll
        for (int ni = 0; ni < 2; ++ni)
            acc[mi][ni] = (v8f)0.0f;

    // Stage one 64x64 f16 slab of A and B into buffer b (8 x b128 per thread).
    auto stage = [&](int b, int kb) {
#pragma unroll
        for (int i = 0; i < 4; ++i) {
            const int e  = (tid + i * 128) * 8;  // half index within 64x64 tile
            const int r  = e >> 6;               // row 0..63
            const int kc = e & 63;               // k, multiple of 8
            copy16_to_lds(fnh + (size_t)(bm + r) * K_ + kb + kc,
                          &As[b][r * LDSROW + kc]);
            copy16_to_lds(fnh + (size_t)(bn + r) * K_ + kb + kc,
                          &Bs[b][r * LDSROW + kc]);
        }
    };

    stage(0, k0);
    async_join();
    __syncthreads();

    const int STEPS = KSLICE / FBK;  // 64
    for (int s = 0; s < STEPS; ++s) {
        const int cur = s & 1;
        if (s + 1 < STEPS) stage(cur ^ 1, k0 + (s + 1) * FBK);  // prefetch next slab

        const _Float16* A = As[cur];
        const _Float16* B = Bs[cur];
#pragma unroll
        for (int kk = 0; kk < FBK; kk += 32) {
            // A fragment (16x32): lane<16 -> K 0..15, lane>=16 -> K 16..31
            v16h a[2];
#pragma unroll
            for (int mi = 0; mi < 2; ++mi) {
                const _Float16* p = A + (wrow * 32 + mi * 16 + lrow) * LDSROW + kk + hsel * 16;
                v8h lo = *(const v8h*)p;
                v8h hi = *(const v8h*)(p + 8);
                a[mi] = __builtin_shufflevector(lo, hi, 0, 1, 2, 3, 4, 5, 6, 7,
                                                8, 9, 10, 11, 12, 13, 14, 15);
            }
            // B fragment (32x16): lane<16 -> K 0..15, lane>=16 -> K 16..31
            v16h b[2];
#pragma unroll
            for (int ni = 0; ni < 2; ++ni) {
                const _Float16* p = B + (wcol * 32 + ni * 16 + lrow) * LDSROW + kk + hsel * 16;
                v8h lo = *(const v8h*)p;
                v8h hi = *(const v8h*)(p + 8);
                b[ni] = __builtin_shufflevector(lo, hi, 0, 1, 2, 3, 4, 5, 6, 7,
                                                8, 9, 10, 11, 12, 13, 14, 15);
            }
#pragma unroll
            for (int mi = 0; mi < 2; ++mi)
#pragma unroll
                for (int ni = 0; ni < 2; ++ni)
                    acc[mi][ni] = __builtin_amdgcn_wmma_f32_16x16x32_f16(
                        false, a[mi], false, b[ni], (short)0, acc[mi][ni],
                        false, false);
        }
        async_join();
        __syncthreads();
    }

    // Raw partial sums (scale + diag mask applied in the reduce kernel).
    float* out = partial + (size_t)blockIdx.z * N_ * N_;
#pragma unroll
    for (int mi = 0; mi < 2; ++mi)
#pragma unroll
        for (int ni = 0; ni < 2; ++ni)
#pragma unroll
            for (int v = 0; v < 8; ++v) {
                const int row = bm + wrow * 32 + mi * 16 + v + hsel * 8;
                const int col = bn + wcol * 32 + ni * 16 + lrow;
                out[row * N_ + col] = acc[mi][ni][v];
            }
}

// ---------------------------------------------------------------------------
// FAST PATH kernel 3: sum SPLITK partials, scale by 1/(T*temp), mask diagonal.
// ---------------------------------------------------------------------------
__global__ __launch_bounds__(256)
void reduce_kernel(const float* __restrict__ partial, float* __restrict__ sim) {
    const int idx = blockIdx.x * 256 + threadIdx.x;  // 0..N*N-1
    float s = 0.0f;
#pragma unroll
    for (int z = 0; z < SPLITK; ++z) s += partial[(size_t)z * N_ * N_ + idx];
    const int row = idx >> 9;
    const int col = idx & (N_ - 1);
    float val = s * (1.0f / ((float)T_ * 0.5f));
    if (row == col) val = -9.0e15f / 0.5f;
    sim[idx] = val;
}

// ---------------------------------------------------------------------------
// FALLBACK kernels (low-workspace path, unchanged from round 1)
// ---------------------------------------------------------------------------
__global__ __launch_bounds__(T_)
void norm_kernel(const float* __restrict__ feats, float* __restrict__ inv_norm) {
    const int n = blockIdx.x;
    const int t = threadIdx.x;
    const float* base = feats + (size_t)n * K_ + t;
    float acc = 0.0f;
#pragma unroll 8
    for (int d = 0; d < D_; ++d) {
        float v = base[(size_t)d * T_];
        acc += v * v;
    }
    inv_norm[n * T_ + t] = 1.0f / fmaxf(sqrtf(acc), 1e-8f);
}

__global__ __launch_bounds__(256)
void gram_kernel(const float* __restrict__ feats,
                 const float* __restrict__ inv_norm,
                 float* __restrict__ sim) {
    __shared__ __align__(16) _Float16 As[BM * LDSROW];
    __shared__ __align__(16) _Float16 Bs[BN * LDSROW];

    const int tid  = threadIdx.x;
    const int lane = tid & 31;
    const int wave = tid >> 5;
    const int wrow = wave >> 1;
    const int wcol = wave & 1;
    const int lrow = lane & 15;
    const int hsel = (lane < 16) ? 0 : 1;
    const int bm = blockIdx.y * BM;
    const int bn = blockIdx.x * BN;

    v8f acc[2][4];
#pragma unroll
    for (int mi = 0; mi < 2; ++mi)
#pragma unroll
        for (int ni = 0; ni < 4; ++ni)
            acc[mi][ni] = (v8f)0.0f;

    for (int kb = 0; kb < K_; kb += BK) {
        __syncthreads();
        const int tbase = kb & (T_ - 1);
#pragma unroll
        for (int i = 0; i < 8; ++i) {
            const int e = (tid + i * 256) * 4;
            const int r = e >> 6;
            const int k = e & 63;
            {
                const int n = bm + r;
                v4f f = *(const v4f*)(feats + (size_t)n * K_ + kb + k);
                v4f w = *(const v4f*)(inv_norm + n * T_ + tbase + k);
                v4h h;
#pragma unroll
                for (int c = 0; c < 4; ++c) h[c] = (_Float16)(f[c] * w[c]);
                *(v4h*)(As + r * LDSROW + k) = h;
            }
            {
                const int n = bn + r;
                v4f f = *(const v4f*)(feats + (size_t)n * K_ + kb + k);
                v4f w = *(const v4f*)(inv_norm + n * T_ + tbase + k);
                v4h h;
#pragma unroll
                for (int c = 0; c < 4; ++c) h[c] = (_Float16)(f[c] * w[c]);
                *(v4h*)(Bs + r * LDSROW + k) = h;
            }
        }
        __syncthreads();

#pragma unroll
        for (int kk = 0; kk < BK; kk += 32) {
            v16h a[2];
#pragma unroll
            for (int mi = 0; mi < 2; ++mi) {
                const _Float16* p = As + (wrow * 32 + mi * 16 + lrow) * LDSROW + kk + hsel * 16;
                v8h lo = *(const v8h*)p;
                v8h hi = *(const v8h*)(p + 8);
                a[mi] = __builtin_shufflevector(lo, hi, 0, 1, 2, 3, 4, 5, 6, 7,
                                                8, 9, 10, 11, 12, 13, 14, 15);
            }
            v16h b[4];
#pragma unroll
            for (int ni = 0; ni < 4; ++ni) {
                const _Float16* p = Bs + (wcol * 64 + ni * 16 + lrow) * LDSROW + kk + hsel * 16;
                v8h lo = *(const v8h*)p;
                v8h hi = *(const v8h*)(p + 8);
                b[ni] = __builtin_shufflevector(lo, hi, 0, 1, 2, 3, 4, 5, 6, 7,
                                                8, 9, 10, 11, 12, 13, 14, 15);
            }
#pragma unroll
            for (int mi = 0; mi < 2; ++mi)
#pragma unroll
                for (int ni = 0; ni < 4; ++ni)
                    acc[mi][ni] = __builtin_amdgcn_wmma_f32_16x16x32_f16(
                        false, a[mi], false, b[ni], (short)0, acc[mi][ni],
                        false, false);
        }
    }

    const float scale = 1.0f / ((float)T_ * 0.5f);
    const float diagv = -9.0e15f / 0.5f;
#pragma unroll
    for (int mi = 0; mi < 2; ++mi)
#pragma unroll
        for (int ni = 0; ni < 4; ++ni)
#pragma unroll
            for (int v = 0; v < 8; ++v) {
                const int row = bm + wrow * 32 + mi * 16 + v + hsel * 8;
                const int col = bn + wcol * 64 + ni * 16 + lrow;
                float val = acc[mi][ni][v] * scale;
                if (row == col) val = diagv;
                sim[row * N_ + col] = val;
            }
}

// ---------------------------------------------------------------------------
// Shared tail: per-row logsumexp minus positive pair, then deterministic mean.
// ---------------------------------------------------------------------------
__global__ __launch_bounds__(128)
void lse_kernel(const float* __restrict__ sim, float* __restrict__ row_nll) {
    const int i = blockIdx.x;
    const int tid = threadIdx.x;
    __shared__ float red[128];
    const float* row = sim + (size_t)i * N_;

    float v[4];
    float m = -3.0e38f;
#pragma unroll
    for (int j = 0; j < 4; ++j) {
        v[j] = row[tid + j * 128];
        m = fmaxf(m, v[j]);
    }
    red[tid] = m;
    __syncthreads();
    for (int s = 64; s > 0; s >>= 1) {
        if (tid < s) red[tid] = fmaxf(red[tid], red[tid + s]);
        __syncthreads();
    }
    const float M = red[0];
    __syncthreads();

    float sum = 0.0f;
#pragma unroll
    for (int j = 0; j < 4; ++j) sum += __expf(v[j] - M);
    red[tid] = sum;
    __syncthreads();
    for (int s = 64; s > 0; s >>= 1) {
        if (tid < s) red[tid] += red[tid + s];
        __syncthreads();
    }
    if (tid == 0) {
        const float lse = M + __logf(red[0]);
        const float pos = row[(i + N_ / 2) & (N_ - 1)];
        row_nll[i] = lse - pos;
    }
}

__global__ __launch_bounds__(256)
void mean_kernel(const float* __restrict__ row_nll, float* __restrict__ out) {
    __shared__ float red[256];
    const int tid = threadIdx.x;
    red[tid] = row_nll[tid] + row_nll[tid + 256];
    __syncthreads();
    for (int s = 128; s > 0; s >>= 1) {
        if (tid < s) red[tid] += red[tid + s];
        __syncthreads();
    }
    if (tid == 0) out[0] = red[0] / (float)N_;
}

// ---------------------------------------------------------------------------
extern "C" void kernel_launch(void* const* d_in, const int* in_sizes, int n_in,
                              void* d_out, int out_size, void* d_ws, size_t ws_size,
                              hipStream_t stream) {
    (void)in_sizes; (void)n_in; (void)out_size;
    const float* feats = (const float*)d_in[0];
    float* out = (float*)d_out;
    char* w = (char*)d_ws;

    const size_t fast_need = ((size_t)43 << 20);  // fnh 32MB + partial 8MB + sim 1MB + nll

    if (ws_size >= fast_need) {
        // Fast path: f16 normalized matrix + split-K WMMA GEMM (512 blocks).
        _Float16* fnh    = (_Float16*)w;                       // 32 MiB
        float* partial   = (float*)(w + ((size_t)32 << 20));   // 8 MiB
        float* sim       = (float*)(w + ((size_t)40 << 20));   // 1 MiB
        float* row_nll   = (float*)(w + ((size_t)41 << 20));   // 2 KiB

        fnh_kernel<<<N_, 256, 0, stream>>>(feats, fnh);
        dim3 grid(N_ / FBN, N_ / FBM, SPLITK);  // 8x8x8
        gram_f16_kernel<<<grid, 128, 0, stream>>>(fnh, partial);
        reduce_kernel<<<(N_ * N_) / 256, 256, 0, stream>>>(partial, sim);
        lse_kernel<<<N_, 128, 0, stream>>>(sim, row_nll);
        mean_kernel<<<1, 256, 0, stream>>>(row_nll, out);
    } else {
        // Fallback: ~1.3 MB workspace, fused conversion in the GEMM.
        float* inv_norm = (float*)w;                    // N*T floats
        float* sim      = inv_norm + (size_t)N_ * T_;   // N*N floats
        float* row_nll  = sim + (size_t)N_ * N_;        // N floats

        norm_kernel<<<N_, T_, 0, stream>>>(feats, inv_norm);
        dim3 grid(N_ / BN, N_ / BM);  // 4x4
        gram_kernel<<<grid, 256, 0, stream>>>(feats, inv_norm, sim);
        lse_kernel<<<N_, 128, 0, stream>>>(sim, row_nll);
        mean_kernel<<<1, 256, 0, stream>>>(row_nll, out);
    }
}
